// MultiGaussKernelEE_28724741276091
// MI455X (gfx1250) — matrix operand order repaired and checked by reading, not compiled, not run on hardware
//
#include <hip/hip_runtime.h>
#include <hip/hip_bf16.h>

#define KCOMP 128
#define DDIM  256
#define NPTS  8192
#define ROWS_PER_GROUP 128
#define GROUPS 4
#define ROWS_PER_WG (ROWS_PER_GROUP * GROUPS)   // 512

typedef __attribute__((ext_vector_type(16))) __bf16 v16bf;
typedef __attribute__((ext_vector_type(8)))  float  v8f;

// ---------------------------------------------------------------------------
// Stage 1: per-component constants  constK[k] = sum_d log|diag[k,d]| + log_softmax(weigh)[k]
// ---------------------------------------------------------------------------
__global__ void gmm_const_kernel(const float* __restrict__ diag,
                                 const float* __restrict__ weigh,
                                 float* __restrict__ constK) {
  __shared__ float sh[KCOMP];
  const int k = threadIdx.x;
  float ld = 0.f;
  for (int d = 0; d < DDIM; ++d) ld += __logf(fabsf(diag[k * DDIM + d]));
  const float wk = weigh[k];
  sh[k] = wk;
  __syncthreads();
  float m = -INFINITY;
  for (int j = 0; j < KCOMP; ++j) m = fmaxf(m, sh[j]);
  float s = 0.f;
  for (int j = 0; j < KCOMP; ++j) s += __expf(sh[j] - m);
  constK[k] = ld + (wk - m - __logf(s));
}

// ---------------------------------------------------------------------------
// Stage 2: main WMMA kernel.
// grid = (NPTS/512, KCOMP), block = 256 threads (8 waves).
// Whole tril_k staged ONCE in 128 KB LDS (bf16, strictly-lower masked,
// pre-swizzled into B-fragment order).  4 sequential row-groups of 128 rows
// amortize the fill.  Per group x chunk: build one A fragment (8 VGPRs),
// then 16 WMMAs with B fragments loaded straight from LDS (32B per lane).
// Loops over chunk/group kept rolled to keep register pressure low (the
// round-2 variant with prebuilt A frags + manual double buffering spilled).
// ---------------------------------------------------------------------------
__launch_bounds__(256)
__global__ void gmm_main_kernel(const float* __restrict__ x,
                                const float* __restrict__ means,
                                const float* __restrict__ diag,
                                const float* __restrict__ tri,
                                const float* __restrict__ constK,
                                float* __restrict__ lp) {
  // ldsB[chunk c][tile t][lane][s]: fragment idx = c*16+t, lane = (d&15) + 16*((e_local>>3)&1),
  // s = (e_local&7) + 8*(e_local>>4) -> 16 bf16 of e per lane, one 32B-aligned v16bf read.
  extern __shared__ __align__(64) __bf16 ldsB[];   // 8*16*32*16 = 64K bf16 = 128 KB

  const int k      = blockIdx.y;
  const int tid    = threadIdx.x;
  const int lane   = tid & 31;
  const int wave   = tid >> 5;
  const int halfHi = lane >> 4;   // 0: lanes 0-15, 1: lanes 16-31
  const int lm     = lane & 15;

  const float* __restrict__ trik = tri   + (size_t)k * DDIM * DDIM;
  const float* __restrict__ mrow = means + (size_t)k * DDIM;
  const float* __restrict__ drow = diag  + (size_t)k * DDIM;

  // ---- Phase 1: stage whole strictly-lower tril_k into LDS, swizzled ----
  // thread tid owns column e = tid; loop over rows d (coalesced across lanes).
  {
    const int e  = tid;
    const int cB = e >> 5;
    const int el = e & 31;
#pragma unroll 4
    for (int d = 0; d < DDIM; ++d) {
      const float v = (e < d) ? trik[d * DDIM + e] : 0.f;
      ldsB[((cB * 16 + (d >> 4)) * 32 + (d & 15) + (((el >> 3) & 1) << 4)) * 16 + (el & 7) + ((el >> 4) << 3)] = (__bf16)v;
    }
  }
  __syncthreads();

  const float ck = constK[k];
  const v8f vzero = {0.f, 0.f, 0.f, 0.f, 0.f, 0.f, 0.f, 0.f};

#pragma unroll 1
  for (int g = 0; g < GROUPS; ++g) {
    const int rowBase = blockIdx.x * ROWS_PER_WG + g * ROWS_PER_GROUP + wave * 16;
    const float* __restrict__ xrow = x + (size_t)(rowBase + lm) * DDIM;

    v8f acc[16];
#pragma unroll
    for (int t = 0; t < 16; ++t) acc[t] = vzero;

#pragma unroll 1
    for (int c = 0; c < 8; ++c) {
      // ---- A fragment for chunk c (16-bit A layout) ----
      // lane<16: K = {0..7,16..23}+32c ; lane>=16: K = {8..15,24..31}+32c ; M = lane&15
      const int ebase = c * 32 + (halfHi << 3);
      const float4 xa = *(const float4*)(xrow + ebase);
      const float4 xb = *(const float4*)(xrow + ebase + 4);
      const float4 xc = *(const float4*)(xrow + ebase + 16);
      const float4 xd = *(const float4*)(xrow + ebase + 20);
      const float4 ma = *(const float4*)(mrow + ebase);
      const float4 mb = *(const float4*)(mrow + ebase + 4);
      const float4 mc = *(const float4*)(mrow + ebase + 16);
      const float4 md = *(const float4*)(mrow + ebase + 20);
      v16bf a;
      a[0]  = (__bf16)(xa.x - ma.x); a[1]  = (__bf16)(xa.y - ma.y);
      a[2]  = (__bf16)(xa.z - ma.z); a[3]  = (__bf16)(xa.w - ma.w);
      a[4]  = (__bf16)(xb.x - mb.x); a[5]  = (__bf16)(xb.y - mb.y);
      a[6]  = (__bf16)(xb.z - mb.z); a[7]  = (__bf16)(xb.w - mb.w);
      a[8]  = (__bf16)(xc.x - mc.x); a[9]  = (__bf16)(xc.y - mc.y);
      a[10] = (__bf16)(xc.z - mc.z); a[11] = (__bf16)(xc.w - mc.w);
      a[12] = (__bf16)(xd.x - md.x); a[13] = (__bf16)(xd.y - md.y);
      a[14] = (__bf16)(xd.z - md.z); a[15] = (__bf16)(xd.w - md.w);

      // ---- 16 d-tiles: B column = d (lane&15), 16 K values per lane ----
      const __bf16* fragBase = ldsB + ((size_t)c * 16 * 32 + lane) * 16;
#pragma unroll
      for (int t = 0; t < 16; ++t) {
        const v16bf b = *(const v16bf*)(fragBase + (size_t)t * 32 * 16);
        acc[t] = __builtin_amdgcn_wmma_f32_16x16x32_bf16(
            /*neg_a=*/false, a, /*neg_b=*/false, b,
            /*c_mod=*/(short)0, acc[t], /*reuse_a=*/false, /*reuse_b=*/false);
      }
    }

    // ---- add diag*y, square, reduce over d ----
    // C layout: VGPR r -> M = r + 8*halfHi, lane&15 -> d within tile.
    float ss[8];
#pragma unroll
    for (int r = 0; r < 8; ++r) ss[r] = 0.f;

#pragma unroll
    for (int t = 0; t < 16; ++t) {
      const int d    = t * 16 + lm;
      const float dg = drow[d];
      const float mv = mrow[d];
#pragma unroll
      for (int r = 0; r < 8; ++r) {
        const int m    = r + halfHi * 8;
        const float yv = x[(size_t)(rowBase + m) * DDIM + d] - mv;
        const float z  = acc[t][r] + dg * yv;
        ss[r] += z * z;
      }
    }

#pragma unroll
    for (int r = 0; r < 8; ++r) {
      float v = ss[r];
      v += __shfl_xor(v, 1, 32);
      v += __shfl_xor(v, 2, 32);
      v += __shfl_xor(v, 4, 32);
      v += __shfl_xor(v, 8, 32);   // full sum within each 16-lane half
      if (lm == 0) {
        const int m = r + halfHi * 8;
        lp[(size_t)(rowBase + m) * KCOMP + k] = -0.5f * v + ck;
      }
    }
  }
}

// ---------------------------------------------------------------------------
// Stage 3: per-point logsumexp over K, |-(logC + lse)|, block partial sums
// ---------------------------------------------------------------------------
__global__ void gmm_lse_kernel(const float* __restrict__ lp,
                               float* __restrict__ partial) {
  __shared__ float sh[256];
  const int n = blockIdx.x * 256 + threadIdx.x;
  const float* __restrict__ row = lp + (size_t)n * KCOMP;
  float m = -INFINITY;
  for (int j = 0; j < KCOMP; ++j) m = fmaxf(m, row[j]);
  float s = 0.f;
  for (int j = 0; j < KCOMP; ++j) s += __expf(row[j] - m);
  const float logC = -128.f * 1.8378770664093453f;   // -(D/2) log(2*pi)
  const float val  = fabsf(logC + m + __logf(s));
  sh[threadIdx.x] = val;
  __syncthreads();
  for (int st = 128; st > 0; st >>= 1) {
    if (threadIdx.x < st) sh[threadIdx.x] += sh[threadIdx.x + st];
    __syncthreads();
  }
  if (threadIdx.x == 0) partial[blockIdx.x] = sh[0];
}

// ---------------------------------------------------------------------------
// Stage 4: final reduce (32 partials) -> mean
// ---------------------------------------------------------------------------
__global__ void gmm_final_kernel(const float* __restrict__ partial,
                                 float* __restrict__ out) {
  float v = partial[threadIdx.x];
  v += __shfl_xor(v, 1, 32);
  v += __shfl_xor(v, 2, 32);
  v += __shfl_xor(v, 4, 32);
  v += __shfl_xor(v, 8, 32);
  v += __shfl_xor(v, 16, 32);
  if (threadIdx.x == 0) out[0] = v * (1.0f / (float)NPTS);
}

// ---------------------------------------------------------------------------
extern "C" void kernel_launch(void* const* d_in, const int* in_sizes, int n_in,
                              void* d_out, int out_size, void* d_ws, size_t ws_size,
                              hipStream_t stream) {
  const float* x     = (const float*)d_in[0];   // [N, D]
  const float* means = (const float*)d_in[1];   // [K, D]
  const float* diag  = (const float*)d_in[2];   // [1, K, D]
  const float* tri   = (const float*)d_in[3];   // [1, K, D, D]
  const float* weigh = (const float*)d_in[4];   // [1, K]

  float* lp      = (float*)d_ws;                     // N*K floats (4 MB)
  float* constK  = lp + (size_t)NPTS * KCOMP;        // K floats
  float* partial = constK + KCOMP;                   // 32 floats

  gmm_const_kernel<<<1, KCOMP, 0, stream>>>(diag, weigh, constK);

  const size_t ldsBytes = (size_t)8 * 16 * 32 * 16 * sizeof(__bf16);  // 128 KB
  gmm_main_kernel<<<dim3(NPTS / ROWS_PER_WG, KCOMP), 256, ldsBytes, stream>>>(
      x, means, diag, tri, constK, lp);

  gmm_lse_kernel<<<NPTS / 256, 256, 0, stream>>>(lp, partial);
  gmm_final_kernel<<<1, 32, 0, stream>>>(partial, (float*)d_out);
}
